// CrossAttentionFusion_27599459844459
// MI455X (gfx1250) — compile-verified, matrix operand order checked by reading, not run on hardware
//
#include <hip/hip_runtime.h>
#include <cstdint>
#include <cstddef>

#define NB 4
#define CH 256
#define C2 512
#define HW 4096
#define NH 4
#define HD 64

typedef __attribute__((ext_vector_type(16))) _Float16 v16h;
typedef __attribute__((ext_vector_type(8)))  float    v8f;

union Frag16 { v16h v; uint32_t u[8]; _Float16 h[16]; };
union FragC  { v8f  v; float f[8]; };

__device__ __forceinline__ v8f wmma_f16(v16h a, v16h b, v8f c) {
  // D = A(16x32 f16) x B(32x16 f16) + C(16x16 f32)
  return __builtin_amdgcn_wmma_f32_16x16x32_f16(
      /*neg_a=*/false, a, /*neg_b=*/false, b,
      /*c_mod=*/(short)0, c, /*reuse_a=*/false, /*reuse_b=*/false);
}

// Load a 16x32 f16 fragment (A layout; B uses the mirrored layout with N in
// place of M). Element (row, k) lives at base[row*stride + k0 + k]; k pairs
// are contiguous, so each VGPR is one aligned 32-bit read.
// Lane mapping (CDNA5 ISA 7.12.2): lane = (row&15) + 16*hi; VGPR t holds
// k = (t<4?0:16) + hi*8 + 2*(t&3) .. +1.
__device__ __forceinline__ v16h load_frag(const _Float16* base, int row,
                                          int stride, int k0, int hi) {
  Frag16 f;
  const _Float16* p = base + (size_t)row * stride + k0 + hi * 8;
#pragma unroll
  for (int t = 0; t < 8; ++t) {
    int off = ((t < 4) ? 0 : 16) + (t & 3) * 2;
    f.u[t] = *reinterpret_cast<const uint32_t*>(p + off);
  }
  return f.v;
}

// ---------------------------------------------------------------------------
// GEMM: Y[256 x 4096] = W[256x256] * X[256x4096] per batch, f16 WMMA.
// STORE_T=1 -> per-head transposed store [b*NH+h][p][d]  (Q, K)
// STORE_T=0 -> channel-major store       [b][oc][p]       (V, proj)
// ---------------------------------------------------------------------------
template <typename TX, int STORE_T>
__global__ __launch_bounds__(128) void k_gemm256(const TX* __restrict__ X,
                                                 const float* __restrict__ W,
                                                 _Float16* __restrict__ Y) {
  __shared__ _Float16 sW[64][32];   // [m][k]
  __shared__ _Float16 sX[64][34];   // [n][k], padded (stride even -> aligned pairs)
  const int b = blockIdx.z;
  const int mBase = blockIdx.y * 64;
  const int nBase = blockIdx.x * 64;
  const int t = threadIdx.x;
  const int wv = t >> 5, lane = t & 31, lo = lane & 15, hi = lane >> 4;
  const TX* Xb = X + (size_t)b * CH * HW;

  v8f acc[4] = {};
#pragma unroll 1
  for (int k0 = 0; k0 < CH; k0 += 32) {
    {
      int r = t >> 1, kh = (t & 1) * 16;
      const float* wp = W + (size_t)(mBase + r) * CH + k0 + kh;
#pragma unroll
      for (int i = 0; i < 16; ++i) sW[r][kh + i] = (_Float16)wp[i];
    }
    {
      int kk = t >> 2, n4 = (t & 3) * 16;
      const TX* xp = Xb + (size_t)(k0 + kk) * HW + nBase + n4;
#pragma unroll
      for (int i = 0; i < 16; ++i) sX[n4 + i][kk] = (_Float16)(float)xp[i];
    }
    __syncthreads();
    v16h aW = load_frag(&sW[0][0], wv * 16 + lo, 32, 0, hi);
#pragma unroll
    for (int nt = 0; nt < 4; ++nt) {
      v16h bX = load_frag(&sX[0][0], nt * 16 + lo, 34, 0, hi);
      acc[nt] = wmma_f16(aW, bX, acc[nt]);
    }
    __syncthreads();
  }

#pragma unroll
  for (int nt = 0; nt < 4; ++nt) {
    FragC c; c.v = acc[nt];
    int n = nBase + nt * 16 + lo;
#pragma unroll
    for (int r = 0; r < 8; ++r) {
      int m = mBase + wv * 16 + r + 8 * hi;   // output channel
      _Float16 val = (_Float16)c.f[r];
      if (STORE_T) {
        int h = m >> 6, d = m & 63;
        Y[(((size_t)(b * NH + h) * HW) + n) * HD + d] = val;
      } else {
        Y[((size_t)(b * CH + m)) * HW + n] = val;
      }
    }
  }
}

// ---------------------------------------------------------------------------
// Flash attention per (b, head), LDS-free.
// Each wave owns 16 query columns and computes S^T tiles (m = key j, n =
// query q). Under the CDNA5 D-layout, lane = q: softmax stats are per-lane
// scalars (one shfl_xor(16) per reduction), and the D-layout of exp(S^T) is
// bit-identical to the B-fragment layout of P^T for O^T = V^T x P^T — the
// repack is pure in-lane cvt, no LDS round-trip.
// Q,K: [bh][4096][64] f16; V: [b][256][4096] channel-major (row = h*64+d).
// ---------------------------------------------------------------------------
__global__ __launch_bounds__(128) void k_attn(const _Float16* __restrict__ Qg,
                                              const _Float16* __restrict__ Kg,
                                              const _Float16* __restrict__ Vg,
                                              _Float16* __restrict__ Og) {
  const int bh = blockIdx.y, b = bh >> 2, h = bh & 3;
  const int t = threadIdx.x, wv = t >> 5, lane = t & 31, lo = lane & 15, hi = lane >> 4;
  const int qBase = blockIdx.x * 64 + wv * 16;
  const _Float16* Q = Qg + (size_t)bh * HW * HD;
  const _Float16* K = Kg + (size_t)bh * HW * HD;
  const _Float16* V = Vg + (size_t)(b * CH + h * HD) * HW;   // [d][p]

  v16h bQ0 = load_frag(Q, qBase + lo, HD, 0, hi);   // B: (k=d, n=q)
  v16h bQ1 = load_frag(Q, qBase + lo, HD, 32, hi);

  float m_run = -3.0e38f, l_run = 0.f;   // per-lane: stats of query column lo
  v8f acc[4] = {};                       // O^T tiles: m = d (4x16), n = q

#pragma unroll 1
  for (int j0 = 0; j0 < HW; j0 += 32) {
    v16h aK0a = load_frag(K, j0 + lo, HD, 0, hi);        // A: (m=j, k=d)
    v16h aK0b = load_frag(K, j0 + lo, HD, 32, hi);
    v16h aK1a = load_frag(K, j0 + 16 + lo, HD, 0, hi);
    v16h aK1b = load_frag(K, j0 + 16 + lo, HD, 32, hi);
    v8f z = {};
    FragC s0, s1;   // S^T tiles: rows j0..j0+15 / j0+16..j0+31, col = q
    s0.v = wmma_f16(aK0b, bQ1, wmma_f16(aK0a, bQ0, z));
    s1.v = wmma_f16(aK1b, bQ1, wmma_f16(aK1a, bQ0, z));

    float mloc = -3.0e38f;
#pragma unroll
    for (int r = 0; r < 8; ++r) {
      s0.f[r] *= 0.125f;               // scale = 1/sqrt(64)
      s1.f[r] *= 0.125f;
      mloc = fmaxf(mloc, fmaxf(s0.f[r], s1.f[r]));
    }
    mloc = fmaxf(mloc, __shfl_xor(mloc, 16, 32));   // join the two j half-groups
    float nmax = fmaxf(m_run, mloc);
    float corr = __expf(m_run - nmax);
    m_run = nmax;

    Frag16 pf;      // P^T as B fragment: halfs 0..7 = tile0 regs, 8..15 = tile1
    float ps = 0.f;
#pragma unroll
    for (int r = 0; r < 8; ++r) {
      float pa = __expf(s0.f[r] - nmax);
      float pb = __expf(s1.f[r] - nmax);
      ps += pa + pb;
      pf.h[r] = (_Float16)pa;
      pf.h[8 + r] = (_Float16)pb;
    }
    ps += __shfl_xor(ps, 16, 32);
    l_run = l_run * corr + ps;

#pragma unroll
    for (int mt = 0; mt < 4; ++mt) {    // rescale running O^T (per-lane scalar)
      FragC cc; cc.v = acc[mt];
#pragma unroll
      for (int r = 0; r < 8; ++r) cc.f[r] *= corr;
      acc[mt] = cc.v;
    }
#pragma unroll
    for (int mt = 0; mt < 4; ++mt) {    // O^T += V^T(16d x 32j) x P^T(32j x 16q)
      v16h aV = load_frag(V, mt * 16 + lo, HW, j0, hi);
      acc[mt] = wmma_f16(aV, pf.v, acc[mt]);
    }
  }

  const float rinv = 1.f / l_run;
#pragma unroll
  for (int mt = 0; mt < 4; ++mt) {
    FragC cc; cc.v = acc[mt];
#pragma unroll
    for (int r = 0; r < 8; ++r) {
      int ch = h * HD + mt * 16 + r + 8 * hi;
      Og[((size_t)(b * CH + ch)) * HW + (qBase + lo)] = (_Float16)(cc.f[r] * rinv);
    }
  }
}

// ---------------------------------------------------------------------------
// Channel attention pooling: avg/max over HW for each of 512 cat channels.
// ---------------------------------------------------------------------------
__global__ void k_chstats(const _Float16* __restrict__ po, const float* __restrict__ gray,
                          float* __restrict__ avg, float* __restrict__ mx) {
  __shared__ float ss[256], sm[256];
  const int b = blockIdx.y, ch = blockIdx.x, t = threadIdx.x;
  float s = 0.f, m = -3.0e38f;
  if (ch < CH) {
    const _Float16* p = po + (size_t)(b * CH + ch) * HW;
    for (int i = t; i < HW; i += 256) { float v = (float)p[i]; s += v; m = fmaxf(m, v); }
  } else {
    const float* p = gray + (size_t)(b * CH + (ch - CH)) * HW;
    for (int i = t; i < HW; i += 256) { float v = p[i]; s += v; m = fmaxf(m, v); }
  }
  ss[t] = s; sm[t] = m; __syncthreads();
  for (int o = 128; o > 0; o >>= 1) {
    if (t < o) { ss[t] += ss[t + o]; sm[t] = fmaxf(sm[t], sm[t + o]); }
    __syncthreads();
  }
  if (t == 0) { avg[b * C2 + ch] = ss[0] * (1.f / HW); mx[b * C2 + ch] = sm[0]; }
}

// SE MLP: wch = sigmoid(W2*relu(W1*avg) + W2*relu(W1*mx)); one block per batch.
__global__ void k_semlp(const float* __restrict__ avg, const float* __restrict__ mx,
                        const float* __restrict__ w1, const float* __restrict__ w2,
                        float* __restrict__ wch) {
  __shared__ float hsum[64];
  const int b = blockIdx.x, t = threadIdx.x;
  const float* a = avg + b * C2;
  const float* m = mx + b * C2;
  float ha = 0.f, hm = 0.f;
  for (int c = 0; c < C2; ++c) { float w = w1[t * C2 + c]; ha += w * a[c]; hm += w * m[c]; }
  hsum[t] = fmaxf(ha, 0.f) + fmaxf(hm, 0.f);
  __syncthreads();
  for (int o = t; o < C2; o += 64) {
    float s = 0.f;
    for (int hh = 0; hh < 64; ++hh) s += w2[o * 64 + hh] * hsum[hh];
    wch[b * C2 + o] = 1.f / (1.f + __expf(-s));
  }
}

// Spatial stats on channel-gated cat: per-pixel mean/max over 512 channels.
__global__ void k_spstats(const _Float16* __restrict__ po, const float* __restrict__ gray,
                          const float* __restrict__ wch, float* __restrict__ camean,
                          float* __restrict__ camax) {
  const int b = blockIdx.y;
  const int p = blockIdx.x * 256 + threadIdx.x;
  const _Float16* pp = po + (size_t)b * CH * HW + p;
  const float* gp = gray + (size_t)b * CH * HW + p;
  const float* wc = wch + b * C2;
  float s = 0.f, m = -3.0e38f;
  for (int c = 0; c < CH; ++c) { float v = (float)pp[(size_t)c * HW] * wc[c]; s += v; m = fmaxf(m, v); }
  for (int c = 0; c < CH; ++c) { float v = gp[(size_t)c * HW] * wc[CH + c];   s += v; m = fmaxf(m, v); }
  camean[b * HW + p] = s * (1.f / C2);
  camax[b * HW + p] = m;
}

// 7x7 spatial-attention conv (2->1 channel), pad 3, sigmoid.
__global__ void k_saconv(const float* __restrict__ camean, const float* __restrict__ camax,
                         const float* __restrict__ sw, float* __restrict__ wsp) {
  const int b = blockIdx.y;
  const int p = blockIdx.x * 256 + threadIdx.x;
  const int y = p >> 6, x = p & 63;
  float acc = 0.f;
  for (int cc = 0; cc < 2; ++cc) {
    const float* src = (cc == 0 ? camean : camax) + b * HW;
    for (int ky = 0; ky < 7; ++ky) {
      int yy = y + ky - 3; if ((unsigned)yy >= 64u) continue;
      for (int kx = 0; kx < 7; ++kx) {
        int xx = x + kx - 3; if ((unsigned)xx >= 64u) continue;
        acc += sw[(cc * 7 + ky) * 7 + kx] * src[yy * 64 + xx];
      }
    }
  }
  wsp[b * HW + p] = 1.f / (1.f + __expf(-acc));
}

// Apply channel + spatial gates, emit f16 cat for the fuse GEMM.
__global__ void k_gate(const _Float16* __restrict__ po, const float* __restrict__ gray,
                       const float* __restrict__ wch, const float* __restrict__ wsp,
                       _Float16* __restrict__ catg) {
  const int b = blockIdx.y, ch = blockIdx.x;
  const float g = wch[b * C2 + ch];
  const float* sp = wsp + b * HW;
  _Float16* dst = catg + (size_t)(b * C2 + ch) * HW;
  if (ch < CH) {
    const _Float16* src = po + (size_t)(b * CH + ch) * HW;
    for (int p = threadIdx.x; p < HW; p += 256) dst[p] = (_Float16)((float)src[p] * g * sp[p]);
  } else {
    const float* src = gray + (size_t)(b * CH + (ch - CH)) * HW;
    for (int p = threadIdx.x; p < HW; p += 256) dst[p] = (_Float16)(src[p] * g * sp[p]);
  }
}

// Repack fuse weights [o][ci][3][3] f32 -> [tap][o][ci] f16 (one shot) so the
// fuse GEMM stages weights with contiguous f16 b128 loads instead of stride-9
// scalar f32 loads on every k-step.
__global__ void k_packfw(const float* __restrict__ fw, _Float16* __restrict__ fwp) {
  int idx = blockIdx.x * 256 + threadIdx.x;
  if (idx >= CH * C2 * 9) return;
  int tap = idx % 9, rem = idx / 9;            // rem = o*512 + ci
  fwp[(size_t)tap * CH * C2 + rem] = (_Float16)fw[idx];
}

// ---------------------------------------------------------------------------
// Fuse 3x3 conv (512->256) as 9 tap-GEMMs over K=512 each, WMMA, then
// BatchNorm (inference) + LeakyReLU(0.2) in f32.
// ---------------------------------------------------------------------------
__global__ __launch_bounds__(128) void k_fuse(const _Float16* __restrict__ catg,
                                              const _Float16* __restrict__ fwp,
                                              const float* __restrict__ bng,
                                              const float* __restrict__ bnb,
                                              const float* __restrict__ bnm,
                                              const float* __restrict__ bnv,
                                              float* __restrict__ out) {
  __shared__ _Float16 sW[64][32];
  __shared__ _Float16 sX[64][34];
  const int b = blockIdx.z, mBase = blockIdx.y * 64, nBase = blockIdx.x * 64;
  const int t = threadIdx.x, wv = t >> 5, lane = t & 31, lo = lane & 15, hi = lane >> 4;
  const _Float16* Xb = catg + (size_t)b * C2 * HW;

  v8f acc[4] = {};
#pragma unroll 1
  for (int tap = 0; tap < 9; ++tap) {
    const int dy = tap / 3 - 1, dx = tap % 3 - 1;
    const _Float16* Wt = fwp + (size_t)tap * CH * C2;
#pragma unroll 1
    for (int k0 = 0; k0 < C2; k0 += 32) {
      {
        int r = t >> 1, kh = (t & 1) * 16;
        const _Float16* wp = Wt + (size_t)(mBase + r) * C2 + k0 + kh;
#pragma unroll
        for (int i = 0; i < 16; ++i) sW[r][kh + i] = wp[i];
      }
      {
        int kk = t >> 2, n4 = (t & 3) * 16;
#pragma unroll
        for (int i = 0; i < 16; ++i) {
          int n = nBase + n4 + i;
          int y = (n >> 6) + dy, x = (n & 63) + dx;
          _Float16 v = (_Float16)0.f;
          if ((unsigned)y < 64u && (unsigned)x < 64u)
            v = Xb[(size_t)(k0 + kk) * HW + y * 64 + x];
          sX[n4 + i][kk] = v;
        }
      }
      __syncthreads();
      v16h aW = load_frag(&sW[0][0], wv * 16 + lo, 32, 0, hi);
#pragma unroll
      for (int nt = 0; nt < 4; ++nt) {
        v16h bX = load_frag(&sX[0][0], nt * 16 + lo, 34, 0, hi);
        acc[nt] = wmma_f16(aW, bX, acc[nt]);
      }
      __syncthreads();
    }
  }

#pragma unroll
  for (int nt = 0; nt < 4; ++nt) {
    FragC cc; cc.v = acc[nt];
    int n = nBase + nt * 16 + lo;
#pragma unroll
    for (int r = 0; r < 8; ++r) {
      int oc = mBase + wv * 16 + r + 8 * hi;
      float inv = bng[oc] * rsqrtf(bnv[oc] + 1e-5f);
      float yv = (cc.f[r] - bnm[oc]) * inv + bnb[oc];
      out[((size_t)(b * CH + oc)) * HW + n] = yv > 0.f ? yv : 0.2f * yv;
    }
  }
}

// ---------------------------------------------------------------------------
extern "C" void kernel_launch(void* const* d_in, const int* in_sizes, int n_in,
                              void* d_out, int out_size, void* d_ws, size_t ws_size,
                              hipStream_t stream) {
  (void)in_sizes; (void)n_in; (void)out_size; (void)ws_size;
  const float* color = (const float*)d_in[0];
  const float* gray  = (const float*)d_in[1];
  const float* Wq    = (const float*)d_in[2];
  const float* Wk    = (const float*)d_in[3];
  const float* Wv    = (const float*)d_in[4];
  const float* Wp    = (const float*)d_in[5];
  const float* caw1  = (const float*)d_in[6];
  const float* caw2  = (const float*)d_in[7];
  const float* saw   = (const float*)d_in[8];
  const float* fw    = (const float*)d_in[9];
  const float* bng   = (const float*)d_in[10];
  const float* bnb   = (const float*)d_in[11];
  const float* bnm   = (const float*)d_in[12];
  const float* bnv   = (const float*)d_in[13];

  char* w = (char*)d_ws;
  auto carve = [&](size_t bytes) {
    char* p = w; w += (bytes + 255) & ~(size_t)255; return p;
  };
  _Float16* qf    = (_Float16*)carve((size_t)NB * NH * HW * HD * 2); // 8 MB
  _Float16* kf    = (_Float16*)carve((size_t)NB * NH * HW * HD * 2); // 8 MB
  _Float16* vf    = (_Float16*)carve((size_t)NB * CH * HW * 2);      // 8 MB
  _Float16* attnf = (_Float16*)carve((size_t)NB * CH * HW * 2);      // 8 MB
  _Float16* projf = (_Float16*)carve((size_t)NB * CH * HW * 2);      // 8 MB
  _Float16* catgf = (_Float16*)carve((size_t)NB * C2 * HW * 2);      // 16 MB
  _Float16* fwp   = (_Float16*)carve((size_t)CH * C2 * 9 * 2);       // 2.25 MB
  float* avgb = (float*)carve((size_t)NB * C2 * 4);
  float* mxb  = (float*)carve((size_t)NB * C2 * 4);
  float* wchb = (float*)carve((size_t)NB * C2 * 4);
  float* cam  = (float*)carve((size_t)NB * HW * 4);
  float* cax  = (float*)carve((size_t)NB * HW * 4);
  float* wspb = (float*)carve((size_t)NB * HW * 4);

  dim3 gg(HW / 64, CH / 64, NB), bg(128);
  k_packfw<<<(CH * C2 * 9 + 255) / 256, 256, 0, stream>>>(fw, fwp);
  k_gemm256<float, 1><<<gg, bg, 0, stream>>>(color, Wq, qf);
  k_gemm256<float, 1><<<gg, bg, 0, stream>>>(gray, Wk, kf);
  k_gemm256<float, 0><<<gg, bg, 0, stream>>>(gray, Wv, vf);
  k_attn<<<dim3(HW / 64, NB * NH), 128, 0, stream>>>(qf, kf, vf, attnf);
  k_gemm256<_Float16, 0><<<gg, bg, 0, stream>>>(attnf, Wp, projf);
  k_chstats<<<dim3(C2, NB), 256, 0, stream>>>(projf, gray, avgb, mxb);
  k_semlp<<<NB, 64, 0, stream>>>(avgb, mxb, caw1, caw2, wchb);
  k_spstats<<<dim3(HW / 256, NB), 256, 0, stream>>>(projf, gray, wchb, cam, cax);
  k_saconv<<<dim3(HW / 256, NB), 256, 0, stream>>>(cam, cax, saw, wspb);
  k_gate<<<dim3(C2, NB), 256, 0, stream>>>(projf, gray, wchb, wspb, catgf);
  k_fuse<<<gg, bg, 0, stream>>>(catgf, fwp, bng, bnb, bnm, bnv, (float*)d_out);
}
